// Raycaster_84799834292671
// MI455X (gfx1250) — compile-verified, matrix operand order checked by reading, not run on hardware
//
#include <hip/hip_runtime.h>

typedef __attribute__((ext_vector_type(16))) _Float16 v16h;
typedef __attribute__((ext_vector_type(8)))  _Float16 v8h;
typedef __attribute__((ext_vector_type(8)))  float    v8f;

#define NUM_RAYS    2048
#define NUM_SAMPLES 128
#define HIDDEN      256
#define STRIDE      264   // LDS row stride in halves: 256 + 8 pad (keeps b128 16B-aligned, spreads banks)
#define KT          8     // 256 / 32  (K tiles per layer)
#define NT          16    // 256 / 16  (N tiles per layer)

// d_ws layout (bytes): packed f16 weight fragments, 1 KB per 16x16x32 fragment
#define W1PACK_OFF 0
#define W2PACK_OFF (128 * 1024)
#define W3PACK_OFF (256 * 1024)

// -------------------------------------------------------------------------
// Pack W1 (256x256), W2 (256x256), W3 (256x4 -> padded N=16) from row-major
// f32 into WMMA B-fragment-native f16 layout:
//   frag (nt,kt): lane l (<16)  holds col n0+l,  K = k0+0..7  | k0+16..23
//                 lane l (>=16) holds col n0+l-16, K = k0+8..15 | k0+24..31
// 16 halves per lane, contiguous (lane stride 32 B).
// -------------------------------------------------------------------------
__global__ void pack_weights(const float* __restrict__ W1,
                             const float* __restrict__ W2,
                             const float* __restrict__ W3,
                             _Float16* __restrict__ ws)
{
    int tid = blockIdx.x * blockDim.x + threadIdx.x;
    const int total = (128 + 128 + 8) * 32;     // fragments * lanes
    if (tid >= total) return;
    int frag = tid >> 5;
    int lane = tid & 31;

    const float* W; int N; long base; int lf;
    if (frag < 128)      { W = W1; N = 256; base = W1PACK_OFF / 2; lf = frag; }
    else if (frag < 256) { W = W2; N = 256; base = W2PACK_OFF / 2; lf = frag - 128; }
    else                 { W = W3; N = 4;   base = W3PACK_OFF / 2; lf = frag - 256; }

    int nt = lf / KT;
    int kt = lf % KT;
    int n  = nt * 16 + (lane & 15);
    int kb = kt * 32 + ((lane < 16) ? 0 : 8);

    _Float16* dst = ws + base + ((long)lf * 32 + lane) * 16;
#pragma unroll
    for (int h = 0; h < 16; ++h) {
        int k = kb + ((h < 8) ? h : (8 + h));   // h<8 -> k0..k0+7 ; h>=8 -> k0+16..k0+23
        float v = (n < N) ? W[k * N + n] : 0.0f;
        dst[h] = (_Float16)v;
    }
}

// -------------------------------------------------------------------------
// One workgroup = one ray (128 samples = 128 MLP rows). 8 waves, each owns a
// 16-row strip end-to-end: no inter-wave data flow until the final blend.
// -------------------------------------------------------------------------
__global__ __launch_bounds__(256)
void nerf_fused_kernel(const float* __restrict__ positions,
                       const float* __restrict__ t_values,
                       const float* __restrict__ W0, const float* __restrict__ b0,
                       const float* __restrict__ b1, const float* __restrict__ b2,
                       const float* __restrict__ b3,
                       const _Float16* __restrict__ wpack,
                       float* __restrict__ out)
{
    __shared__ __align__(16) _Float16 hbuf[NUM_SAMPLES * STRIDE];   // 67584 B
    __shared__ float rgba[NUM_SAMPLES * 4];                         //  2048 B

    const int ray  = blockIdx.x;
    const int lane = threadIdx.x & 31;
    const int wav  = threadIdx.x >> 5;          // 0..7
    const int m0   = wav * 16;                  // wave's row strip

    // ---------------- layer 0: relu(x @ W0 + b0), K=3, plain VALU ----------
    const float* pos = positions + (long)ray * NUM_SAMPLES * 3;
#pragma unroll 1
    for (int cc = 0; cc < 8; ++cc) {
        int col = cc * 32 + lane;
        float w0c = W0[col], w1c = W0[256 + col], w2c = W0[512 + col], bc = b0[col];
#pragma unroll
        for (int r = 0; r < 16; ++r) {
            int row = m0 + r;
            float x0 = pos[row * 3 + 0], x1 = pos[row * 3 + 1], x2 = pos[row * 3 + 2];
            float h = fmaf(x0, w0c, fmaf(x1, w1c, fmaf(x2, w2c, bc)));
            hbuf[row * STRIDE + col] = (_Float16)fmaxf(h, 0.0f);
        }
    }
    // wave-local rows only -> no barrier needed

    const int arow = m0 + (lane & 15);
    const int koff = (lane < 16) ? 0 : 8;

    // ---------------- layers 1 & 2: 256x256 via WMMA -----------------------
#pragma unroll 1
    for (int layer = 0; layer < 2; ++layer) {
        const _Float16* wp   = wpack + ((layer == 0) ? (W1PACK_OFF / 2) : (W2PACK_OFF / 2));
        const float*    bias = (layer == 0) ? b1 : b2;

        // preload the wave's full 16x256 A strip as 8 k-fragments (64 VGPRs)
        v16h a[KT];
#pragma unroll
        for (int kt = 0; kt < KT; ++kt) {
            v8h lo = *(const v8h*)&hbuf[arow * STRIDE + kt * 32 + koff];
            v8h hi = *(const v8h*)&hbuf[arow * STRIDE + kt * 32 + koff + 16];
#pragma unroll
            for (int i = 0; i < 8; ++i) { a[kt][i] = lo[i]; a[kt][8 + i] = hi[i]; }
        }

#pragma unroll 1
        for (int nt = 0; nt < NT; ++nt) {
            v8f acc = {};
#pragma unroll
            for (int kt = 0; kt < KT; ++kt) {
                const _Float16* bp = wp + ((long)(nt * KT + kt) * 32 + lane) * 16;
                v8h blo = *(const v8h*)bp;
                v8h bhi = *(const v8h*)(bp + 8);
                v16h b;
#pragma unroll
                for (int i = 0; i < 8; ++i) { b[i] = blo[i]; b[8 + i] = bhi[i]; }
                acc = __builtin_amdgcn_wmma_f32_16x16x32_f16(
                          false, a[kt], false, b, (short)0, acc, false, false);
            }
            // bias + relu + f16 store back to the wave's own rows (input already in regs)
            int col   = nt * 16 + (lane & 15);
            float bc  = bias[col];
            int rbase = m0 + ((lane < 16) ? 0 : 8);
#pragma unroll
            for (int v = 0; v < 8; ++v) {
                float val = fmaxf(acc[v] + bc, 0.0f);
                hbuf[(rbase + v) * STRIDE + col] = (_Float16)val;
            }
        }
    }

    // ---------------- layer 3: 256 -> 4 (N padded to 16) -------------------
    {
        const _Float16* wp = wpack + W3PACK_OFF / 2;
        v16h a[KT];
#pragma unroll
        for (int kt = 0; kt < KT; ++kt) {
            v8h lo = *(const v8h*)&hbuf[arow * STRIDE + kt * 32 + koff];
            v8h hi = *(const v8h*)&hbuf[arow * STRIDE + kt * 32 + koff + 16];
#pragma unroll
            for (int i = 0; i < 8; ++i) { a[kt][i] = lo[i]; a[kt][8 + i] = hi[i]; }
        }
        v8f acc = {};
#pragma unroll
        for (int kt = 0; kt < KT; ++kt) {
            const _Float16* bp = wp + ((long)kt * 32 + lane) * 16;
            v8h blo = *(const v8h*)bp;
            v8h bhi = *(const v8h*)(bp + 8);
            v16h b;
#pragma unroll
            for (int i = 0; i < 8; ++i) { b[i] = blo[i]; b[8 + i] = bhi[i]; }
            acc = __builtin_amdgcn_wmma_f32_16x16x32_f16(
                      false, a[kt], false, b, (short)0, acc, false, false);
        }
        int col   = lane & 15;
        int rbase = m0 + ((lane < 16) ? 0 : 8);
        if (col < 4) {
            float bc = b3[col];
#pragma unroll
            for (int v = 0; v < 8; ++v) {
                float x = acc[v] + bc;
                float val;
                if (col < 3) val = 1.0f / (1.0f + __expf(-x));                     // sigmoid
                else         val = fmaxf(x, 0.0f) + log1pf(__expf(-fabsf(x)));     // softplus
                rgba[(rbase + v) * 4 + col] = val;
            }
        }
    }

    __syncthreads();

    // ---------------- alpha compositing (serial scan, trivial cost) --------
    if (threadIdx.x == 0) {
        const float* tv = t_values + (long)ray * NUM_SAMPLES;
        float T = 1.0f, c0 = 0.f, c1 = 0.f, c2 = 0.f, asum = 0.f, best = -1.0f;
        int bidx = 0;
#pragma unroll 1
        for (int i = 0; i < NUM_SAMPLES; ++i) {
            float delta = (i < NUM_SAMPLES - 1) ? (tv[i + 1] - tv[i]) : 1e10f;
            float alpha = 1.0f - __expf(-rgba[i * 4 + 3] * delta);
            float wgt   = alpha * T;
            c0 += wgt * rgba[i * 4 + 0];
            c1 += wgt * rgba[i * 4 + 1];
            c2 += wgt * rgba[i * 4 + 2];
            if (i < NUM_SAMPLES - 1) {
                asum += wgt;
                if (wgt > best) { best = wgt; bidx = i; }   // first max (strict >)
            }
            T *= fminf(1.0f, 1.0f - alpha + 1e-10f);
        }
        if (asum < 0.1f) bidx = NUM_SAMPLES - 1;
        out[ray * 3 + 0] = c0;
        out[ray * 3 + 1] = c1;
        out[ray * 3 + 2] = c2;
        out[NUM_RAYS * 3 + ray]            = asum;
        out[NUM_RAYS * 3 + NUM_RAYS + ray] = tv[bidx];
    }
}

extern "C" void kernel_launch(void* const* d_in, const int* in_sizes, int n_in,
                              void* d_out, int out_size, void* d_ws, size_t ws_size,
                              hipStream_t stream) {
    const float* positions = (const float*)d_in[0];
    const float* t_values  = (const float*)d_in[1];
    const float* W0 = (const float*)d_in[2];
    const float* b0 = (const float*)d_in[3];
    const float* W1 = (const float*)d_in[4];
    const float* b1 = (const float*)d_in[5];
    const float* W2 = (const float*)d_in[6];
    const float* b2 = (const float*)d_in[7];
    const float* W3 = (const float*)d_in[8];
    const float* b3 = (const float*)d_in[9];
    _Float16* ws = (_Float16*)d_ws;

    // repack weights into WMMA fragment layout (264 fragments * 32 lanes)
    pack_weights<<<33, 256, 0, stream>>>(W1, W2, W3, ws);

    // one block per ray: fused MLP + compositing
    nerf_fused_kernel<<<NUM_RAYS, 256, 0, stream>>>(
        positions, t_values, W0, b0, b1, b2, b3, ws, (float*)d_out);
}
